// HGSF_84482006712934
// MI455X (gfx1250) — matrix operand order repaired and checked by reading, compiled, not run on hardware
//
#include <hip/hip_runtime.h>
#include <hip/hip_bf16.h>

// Problem constants
#define NF   64      // frames = B*T
#define CC   256     // channels
#define HH   56
#define WW   56
#define HWP  3136    // 56*56
#define HP   58      // padded H
#define WPD  58      // padded W

typedef __attribute__((ext_vector_type(16))) __bf16 v16bf;
typedef __attribute__((ext_vector_type(8)))  float  v8f;

union FragBF { uint4 q[2]; v16bf v; };

__device__ __forceinline__ unsigned short f2bf(float f) {
    unsigned int u = __float_as_uint(f);
    u += 0x7fffu + ((u >> 16) & 1u);   // round-to-nearest-even
    return (unsigned short)(u >> 16);
}

// ---------------------------------------------------------------- zero ws
__global__ void k_zero(uint4* __restrict__ p, long n) {
    long i = (long)blockIdx.x * blockDim.x + threadIdx.x;
    long stride = (long)gridDim.x * blockDim.x;
    uint4 z = make_uint4(0u, 0u, 0u, 0u);
    for (; i < n; i += stride) p[i] = z;
}

// ------------------------------------------------- per-channel BN stats
__global__ void k_stats(const float* __restrict__ x, const float* __restrict__ gamma,
                        const float* __restrict__ beta, float* __restrict__ scale,
                        float* __restrict__ shift) {
    __shared__ float rs[256], rq[256];
    int c = blockIdx.x;
    float s = 0.f, s2 = 0.f;
    for (int i = threadIdx.x; i < NF * HWP; i += 256) {
        int fr = i / HWP, p = i % HWP;
        float v = x[((long)fr * CC + c) * HWP + p];
        s += v; s2 += v * v;
    }
    rs[threadIdx.x] = s; rq[threadIdx.x] = s2;
    __syncthreads();
    for (int st = 128; st > 0; st >>= 1) {
        if (threadIdx.x < st) {
            rs[threadIdx.x] += rs[threadIdx.x + st];
            rq[threadIdx.x] += rq[threadIdx.x + st];
        }
        __syncthreads();
    }
    if (threadIdx.x == 0) {
        const float inv = 1.f / (float)(NF * HWP);
        float mean = rs[0] * inv;
        float var  = rq[0] * inv - mean * mean;
        float sc = gamma[c] * rsqrtf(var + 1e-5f);
        scale[c] = sc;
        shift[c] = beta[c] - mean * sc;
    }
}

// -------------------------------- BN + ReLU -> bf16 NHWC padded [64][58][58][256]
__global__ void k_bnpack(const float* __restrict__ x, const float* __restrict__ scale,
                         const float* __restrict__ shift, unsigned short* __restrict__ xrp) {
    int id = blockIdx.x * 256 + threadIdx.x;   // 0 .. 200703 exactly
    int f  = id / HWP;
    int hw = id % HWP;
    int h = hw / WW, w = hw % WW;
    unsigned short* dst = xrp + ((long)(f * HP + h + 1) * WPD + (w + 1)) * CC;
    for (int ch = 0; ch < 32; ++ch) {
        unsigned int dq[4];
#pragma unroll
        for (int j = 0; j < 4; ++j) {
            int c0 = ch * 8 + j * 2;
            float v0 = x[((long)f * CC + c0) * HWP + hw];
            float v1 = x[((long)f * CC + c0 + 1) * HWP + hw];
            v0 = fmaxf(v0 * scale[c0] + shift[c0], 0.f);
            v1 = fmaxf(v1 * scale[c0 + 1] + shift[c0 + 1], 0.f);
            dq[j] = (unsigned int)f2bf(v0) | ((unsigned int)f2bf(v1) << 16);
        }
        *(uint4*)(dst + ch * 8) = make_uint4(dq[0], dq[1], dq[2], dq[3]);
    }
}

// ------------------- pack gate weights into per-lane WMMA B fragments
// Layout: bpack[g][tap(9)][kc(4)][lane(32)][16 bf16]
// B (32x16): lane<16: N=lane, K={0..7,16..23} ; lane>=16: N=lane-16, K={8..15,24..31}
__global__ void k_packw(const float* __restrict__ gw, unsigned short* __restrict__ bpack) {
    int gid = blockIdx.x * 256 + threadIdx.x;   // 0..2303
    if (gid >= 2 * 9 * 4 * 32) return;
    int lane = gid & 31;
    int kc   = (gid >> 5) & 3;
    int tap  = (gid >> 7) % 9;
    int g    = gid / (32 * 4 * 9);
    int kh = tap / 3, kw = tap % 3;
    int n    = lane & 15;
    int half = lane >> 4;
    unsigned short* dst = bpack + (long)gid * 16;
    for (int e = 0; e < 16; ++e) {
        int K = (e < 8) ? (half * 8 + e) : (16 + half * 8 + (e - 8));
        int c = kc * 32 + K;           // within-group channel
        float v = 0.f;
        if (n < 9) {                   // column n = dd*3 + kt
            int dd = n / 3, kt = n % 3;
            v = gw[(((((long)dd * 2 + g) * 128 + c) * 3 + kt) * 3 + kh) * 3 + kw];
        }
        dst[e] = f2bf(v);
    }
}

// ----------------- implicit-GEMM 3x3 conv2d via WMMA bf16 -> G[64][18][3136]
// WG = (frame, h, group); 4 waves = w-tiles {0,16,32,40}
__global__ void __launch_bounds__(128) k_conv(const unsigned short* __restrict__ xrp,
                                              const unsigned short* __restrict__ bpack,
                                              float* __restrict__ G) {
    __shared__ unsigned short sh[3 * WPD * 128];   // 44,544 B: 3 padded rows x 128 ch
    int bid = blockIdx.x;
    int g = bid & 1;
    int h = (bid >> 1) % HH;
    int f = bid / (2 * HH);

    const int nq = 3 * WPD * 128 / 8;   // 2784 uint4
    for (int i = threadIdx.x; i < nq; i += 128) {
        int r   = i / (WPD * 16);
        int rem = i % (WPD * 16);
        int wp = rem >> 4;
        int cq = rem & 15;
        const uint4* src = (const uint4*)(xrp +
            (((long)(f * HP + h + r) * WPD + wp) * CC + g * 128 + cq * 8));
        *(uint4*)(&sh[(r * WPD + wp) * 128 + cq * 8]) = *src;
    }
    __syncthreads();

    int lane = threadIdx.x & 31;
    int wt   = threadIdx.x >> 5;
    int w0   = (wt < 3) ? wt * 16 : 40;
    int m    = lane & 15;
    int half = lane >> 4;

    v8f acc = {0.f, 0.f, 0.f, 0.f, 0.f, 0.f, 0.f, 0.f};
    const uint4* bp = (const uint4*)(bpack + (long)g * 9 * 4 * 32 * 16);

    for (int kh = 0; kh < 3; ++kh)
        for (int kw = 0; kw < 3; ++kw) {
            int tap = kh * 3 + kw;
            int abase = (kh * WPD + (w0 + m + kw)) * 128 + half * 8;
#pragma unroll
            for (int kc = 0; kc < 4; ++kc) {
                FragBF A, Bf;
                // A (16x32 bf16): lanes<16 K={0..7,16..23}, lanes>=16 K={8..15,24..31}, M=lane&15
                A.q[0] = *(const uint4*)(&sh[abase + kc * 32]);
                A.q[1] = *(const uint4*)(&sh[abase + kc * 32 + 16]);
                const uint4* bpp = bp + ((long)(tap * 4 + kc) * 32 + lane) * 2;
                Bf.q[0] = bpp[0];
                Bf.q[1] = bpp[1];
                acc = __builtin_amdgcn_wmma_f32_16x16x32_bf16(
                    false, A.v, false, Bf.v, (short)0, acc, false, false);
            }
        }

    // D (16x16 f32): lanes<16: N=lane, M=r ; lanes>=16: N=lane-16, M=r+8
    int n = lane & 15;
    if (n < 9) {
        float* dst = G + (long)(f * 18 + g * 9 + n) * HWP + h * WW + w0 + half * 8;
        *(float4*)dst       = make_float4(acc[0], acc[1], acc[2], acc[3]);
        *(float4*)(dst + 4) = make_float4(acc[4], acc[5], acc[6], acc[7]);
    }
}

// --------------- temporal combine + bias + tanh -> gate[3][2][64][3136]
__global__ void k_gate(const float* __restrict__ G, const float* __restrict__ gb,
                       float* __restrict__ gate) {
    int id = blockIdx.x * 256 + threadIdx.x;   // 1,204,224 exactly
    int p    = id % HWP;
    int fr   = (id / HWP) & 63;
    int rest = id / (HWP * 64);
    int g  = rest & 1;
    int dd = rest >> 1;
    int t = fr & 7, b = fr >> 3;
    int d = dd + 1;
    float s = gb[dd * 2 + g];
#pragma unroll
    for (int kt = 0; kt < 3; ++kt) {
        int ts = t + d * (kt - 1);
        if ((unsigned)ts < 8u)
            s += G[(long)((b * 8 + ts) * 18 + g * 9 + dd * 3 + kt) * HWP + p];
    }
    gate[(long)((dd * 2 + g) * 64 + fr) * HWP + p] = tanhf(s);
}

// ------------- fused H,W pooling of x and gate*x for all 3 dilations
__global__ void k_pool(const float* __restrict__ x, const float* __restrict__ gate,
                       float* __restrict__ mx, float* __restrict__ py) {
    __shared__ float4 red[256];
    int bid = blockIdx.x;          // (b*256 + c)*8 + t
    int t = bid & 7;
    int c = (bid >> 3) & 255;
    int b = bid >> 11;
    int fr = b * 8 + t;
    int g = c >> 7;
    const float* xp = x + ((long)fr * CC + c) * HWP;
    const float* g0 = gate + (long)((0 * 2 + g) * 64 + fr) * HWP;
    const float* g1 = gate + (long)((1 * 2 + g) * 64 + fr) * HWP;
    const float* g2 = gate + (long)((2 * 2 + g) * 64 + fr) * HWP;
    float sx = 0.f, s0 = 0.f, s1 = 0.f, s2 = 0.f;
    for (int p = threadIdx.x; p < HWP; p += 256) {
        float xv = xp[p];
        sx += xv;
        s0 += g0[p] * xv;
        s1 += g1[p] * xv;
        s2 += g2[p] * xv;
    }
    red[threadIdx.x] = make_float4(sx, s0, s1, s2);
    __syncthreads();
    for (int st = 128; st > 0; st >>= 1) {
        if (threadIdx.x < st) {
            float4 a = red[threadIdx.x], bq = red[threadIdx.x + st];
            red[threadIdx.x] = make_float4(a.x + bq.x, a.y + bq.y, a.z + bq.z, a.w + bq.w);
        }
        __syncthreads();
    }
    if (threadIdx.x == 0) {
        const float inv = 1.f / 3136.f;
        float4 r = red[0];
        mx[(b * 256 + c) * 8 + t]                 = r.x * inv;
        py[((0 * 8 + b) * 256 + c) * 8 + t]       = r.y * inv;
        py[((1 * 8 + b) * 256 + c) * 8 + t]       = r.z * inv;
        py[((2 * 8 + b) * 256 + c) * 8 + t]       = r.w * inv;
    }
}

// --------- tiny conv2d (2in,1out,3x3 over (ch=128, t=8)) + sigmoid
__global__ void k_wgt(const float* __restrict__ py, const float* __restrict__ mx,
                      const float* __restrict__ c1w, const float* __restrict__ c1b,
                      const float* __restrict__ c2w, const float* __restrict__ c2b,
                      float* __restrict__ wgt) {
    int id = blockIdx.x * 256 + threadIdx.x;   // 49,152: (((dd*2+g)*8+b)*128+cs)*8+t
    int t  = id & 7;
    int cs = (id >> 3) & 127;
    int b  = (id >> 10) & 7;
    int g  = (id >> 13) & 1;
    int dd = id >> 14;
    int d = dd + 1;
    const float* cw = (g == 0 ? c1w : c2w) + dd * 18;
    float s = (g == 0 ? c1b : c2b)[dd];
    for (int kh = 0; kh < 3; ++kh) {
        int ci = cs + kh - 1;
        if ((unsigned)ci > 127u) continue;
        for (int kw = 0; kw < 3; ++kw) {
            int tt = t + kw - 1;
            if ((unsigned)tt > 7u) continue;
            float sy;
            if (g == 0) { int ts = tt + d; sy = (ts < 8)  ? py[((dd * 8 + b) * 256 + ci) * 8 + ts] : 0.f; }
            else        { int ts = tt - d; sy = (ts >= 0) ? py[((dd * 8 + b) * 256 + 128 + ci) * 8 + ts] : 0.f; }
            int cc = g * 128 + ci;
            float pr = mx[(b * 256 + cc) * 8 + tt] - py[((dd * 8 + b) * 256 + cc) * 8 + tt];
            s += cw[kh * 3 + kw] * sy + cw[9 + kh * 3 + kw] * pr;
        }
    }
    wgt[id] = 1.f / (1.f + expf(-s));
}

// ------------- final recombine + temporal shifts + channel shuffle + dilation sum
__global__ void k_final(const float* __restrict__ x, const float* __restrict__ gate,
                        const float* __restrict__ wgt, float* __restrict__ out) {
    long id = (long)blockIdx.x * 256 + threadIdx.x;   // 6,422,528 exactly
    int p = (int)(id % HWP);
    int c = (int)((id / HWP) & 255);
    int b = (int)(id / ((long)HWP * 256));
    int g = c >> 7, cs = c & 127;
    float xv[8], acc[8];
#pragma unroll
    for (int t = 0; t < 8; ++t) {
        xv[t]  = x[((long)((b * 8 + t) * CC + c)) * HWP + p];
        acc[t] = 0.f;
    }
#pragma unroll
    for (int dd = 0; dd < 3; ++dd) {
        int d = dd + 1;
        float gv[8], wv[8];
#pragma unroll
        for (int t = 0; t < 8; ++t) {
            gv[t] = gate[((long)((dd * 2 + g) * 64 + b * 8 + t)) * HWP + p];
            wv[t] = wgt[(((dd * 2 + g) * 8 + b) * 128 + cs) * 8 + t];
        }
#pragma unroll
        for (int t = 0; t < 8; ++t) {
            float y = gv[t] * xv[t];
            float r = xv[t] - y;
            float ys;
            if (g == 0) { int ts = t + d; ys = (ts < 8)  ? gv[ts] * xv[ts] : 0.f; }
            else        { int ts = t - d; ys = (ts >= 0) ? gv[ts] * xv[ts] : 0.f; }
            acc[t] += ys * wv[t] + r * (1.f - wv[t]);
        }
    }
    int nc = (c & 128) | (((cs & 63) << 1) | (cs >> 6));   // channel shuffle
#pragma unroll
    for (int t = 0; t < 8; ++t)
        out[((long)((b * 8 + t) * CC + nc)) * HWP + p] = acc[t];
}

extern "C" void kernel_launch(void* const* d_in, const int* in_sizes, int n_in,
                              void* d_out, int out_size, void* d_ws, size_t ws_size,
                              hipStream_t stream) {
    (void)in_sizes; (void)n_in; (void)out_size; (void)ws_size;
    const float* x     = (const float*)d_in[0];
    const float* gamma = (const float*)d_in[1];
    const float* beta  = (const float*)d_in[2];
    const float* gw    = (const float*)d_in[3];
    const float* gb    = (const float*)d_in[4];
    const float* c1w   = (const float*)d_in[5];
    const float* c1b   = (const float*)d_in[6];
    const float* c2w   = (const float*)d_in[7];
    const float* c2b   = (const float*)d_in[8];
    float* out = (float*)d_out;
    char* ws = (char*)d_ws;

    size_t off = 0;
    unsigned short* xrp = (unsigned short*)(ws + off); off += (size_t)64 * 58 * 58 * 256 * 2; // 110.2 MB
    float* G     = (float*)(ws + off); off += (size_t)64 * 18 * 3136 * 4;   // 14.45 MB
    float* gate  = (float*)(ws + off); off += (size_t)6 * 64 * 3136 * 4;    // 4.82 MB
    float* scale = (float*)(ws + off); off += 1024;
    float* shift = (float*)(ws + off); off += 1024;
    float* mx    = (float*)(ws + off); off += (size_t)8 * 256 * 8 * 4;
    float* py    = (float*)(ws + off); off += (size_t)3 * 8 * 256 * 8 * 4;
    float* wgt   = (float*)(ws + off); off += (size_t)3 * 2 * 8 * 128 * 8 * 4;
    unsigned short* bpack = (unsigned short*)(ws + off); off += (size_t)2 * 9 * 4 * 32 * 16 * 2;

    long nq = (long)64 * 58 * 58 * 256 * 2 / 16;
    k_zero  <<<4096, 256, 0, stream>>>((uint4*)xrp, nq);
    k_stats <<<256, 256, 0, stream>>>(x, gamma, beta, scale, shift);
    k_bnpack<<<784, 256, 0, stream>>>(x, scale, shift, xrp);
    k_packw <<<9, 256, 0, stream>>>(gw, bpack);
    k_conv  <<<64 * 56 * 2, 128, 0, stream>>>(xrp, bpack, G);
    k_gate  <<<4704, 256, 0, stream>>>(G, gb, gate);
    k_pool  <<<16384, 256, 0, stream>>>(x, gate, mx, py);
    k_wgt   <<<192, 256, 0, stream>>>(py, mx, c1w, c1b, c2w, c2b, wgt);
    k_final <<<25088, 256, 0, stream>>>(x, gate, wgt, out);
}
